// SetBankAttention_88003879895287
// MI455X (gfx1250) — compile-verified, matrix operand order checked by reading, not run on hardware
//
#include <hip/hip_runtime.h>
#include <hip/hip_bf16.h>

// ---------------------------------------------------------------------------
// Fused set-bank attention for gfx1250 (MI455X), wave32 + WMMA f16.
// One workgroup = one (batch, head). 256 threads = 8 waves.
// ---------------------------------------------------------------------------

typedef __attribute__((ext_vector_type(16))) _Float16 v16h;
typedef __attribute__((ext_vector_type(8)))  _Float16 v8h;
typedef __attribute__((ext_vector_type(4)))  _Float16 v4h;
typedef __attribute__((ext_vector_type(8)))  float    v8f;

#define D_MODEL 256
#define NHEAD   4
#define DH      64
#define S_SIG   16
#define BATCH   512
#define LQ      128
#define LK      256

// GAMMA*ETA/TAU
#define GE      0.3f
// BETA/(TAU*sqrt(DH))
#define QSCALE  0.125f

// ----- dynamic LDS layout (bytes) -----
#define OFF_PH  0                     // phi staging / P matrix : 128x256 f16 = 65536
#define OFF_W1  65536                 // W slice A/B            : 64x256  f16 = 32768
#define OFF_W2  98304                 // W slice V              : 64x256  f16 = 32768
#define OFF_QB  131072                // q (scaled)             : 128x64  f16 = 16384
#define OFF_KB  147456                // k                      : 256x64  f16 = 32768
#define OFF_VT  180224                // v transposed           : 64x256  f16 = 32768
#define OFF_SQ  212992                // sig_q padded K=32      : 128x32  f16 = 8192
#define OFF_SK  221184                // sig_k padded K=32      : 256x32  f16 = 16384
#define OFF_NQ  237568                // |sig_q|^2              : 128 f32
#define OFF_RQ  238080                // rsqrt(size_q)          : 128 f32
#define OFF_NK  238592                // |sig_k|^2              : 256 f32
#define OFF_RK  239616                // rsqrt(size_k)          : 256 f32
#define SMEM_BYTES 240640

__device__ __forceinline__ v8f wmma_f16(v16h a, v16h b, v8f c) {
    return __builtin_amdgcn_wmma_f32_16x16x32_f16(false, a, false, b, (short)0, c,
                                                  false, false);
}

// Load a 16x32 (MxK) A-frag or 32x16 (KxN) B-frag from row-major f16 LDS.
// Both operand layouts are K-contiguous per lane:
//   lane L -> row/col (row0 + L%16); K base = col0 + 8*(L/16) and +16.
__device__ __forceinline__ v16h ldsFrag(const _Float16* base, int stride,
                                        int row0, int col0, int lane) {
    int l16 = lane & 15, lh = lane >> 4;
    const _Float16* p = base + (row0 + l16) * stride + col0 + 8 * lh;
    v8h a = *(const v8h*)p;
    v8h b = *(const v8h*)(p + 16);
    v16h r;
#pragma unroll
    for (int i = 0; i < 8; ++i) { r[i] = a[i]; r[i + 8] = b[i]; }
    return r;
}

__device__ __forceinline__ v4h pack4(float4 v) {
    v4h h;
    h[0] = (_Float16)v.x; h[1] = (_Float16)v.y;
    h[2] = (_Float16)v.z; h[3] = (_Float16)v.w;
    return h;
}

__global__ __launch_bounds__(256, 1)
void setbank_attn_kernel(const float* __restrict__ phi_q,
                         const float* __restrict__ sig_q,
                         const float* __restrict__ size_q,
                         const float* __restrict__ phi_k,
                         const float* __restrict__ sig_k,
                         const float* __restrict__ size_k,
                         const float* __restrict__ W_A,
                         const float* __restrict__ W_B,
                         const float* __restrict__ W_V,
                         float* __restrict__ out) {
    extern __shared__ __align__(16) char smem[];
    _Float16* PH  = (_Float16*)(smem + OFF_PH);
    _Float16* W1  = (_Float16*)(smem + OFF_W1);
    _Float16* W2  = (_Float16*)(smem + OFF_W2);
    _Float16* QB  = (_Float16*)(smem + OFF_QB);
    _Float16* KB  = (_Float16*)(smem + OFF_KB);
    _Float16* VT  = (_Float16*)(smem + OFF_VT);
    _Float16* SQh = (_Float16*)(smem + OFF_SQ);
    _Float16* SKh = (_Float16*)(smem + OFF_SK);
    float* NQ = (float*)(smem + OFF_NQ);
    float* RQ = (float*)(smem + OFF_RQ);
    float* NK = (float*)(smem + OFF_NK);
    float* RK = (float*)(smem + OFF_RK);

    const int b    = blockIdx.x >> 2;
    const int h    = blockIdx.x & 3;
    const int tid  = threadIdx.x;
    const int wave = tid >> 5;
    const int lane = tid & 31;
    const int l16  = lane & 15, lh = lane >> 4;
    const int m0   = 16 * wave;              // query-row strip of this wave

    // ---------------- Phase 1: stage phi_q, W_A, signatures, norms ----------
    {
        const float4* gp = (const float4*)phi_q + (size_t)b * LQ * (D_MODEL / 4);
        v4h* dp = (v4h*)PH;
        for (int idx = tid; idx < LQ * (D_MODEL / 4); idx += 256)
            dp[idx] = pack4(gp[idx]);

        const float4* gw = (const float4*)W_A + (size_t)h * DH * (D_MODEL / 4);
        v4h* dw = (v4h*)W1;
        for (int idx = tid; idx < DH * (D_MODEL / 4); idx += 256)
            dw[idx] = pack4(gw[idx]);

        const float4* gsq = (const float4*)sig_q + (size_t)b * LQ * (S_SIG / 4);
        for (int idx = tid; idx < LQ * (S_SIG / 4); idx += 256) {
            int row = idx >> 2, c4 = idx & 3;
            *(v4h*)(SQh + row * 32 + c4 * 4)      = pack4(gsq[idx]);
            *(v4h*)(SQh + row * 32 + 16 + c4 * 4) = (v4h){};  // K pad -> 0
        }
        const float4* gsk = (const float4*)sig_k + (size_t)b * LK * (S_SIG / 4);
        for (int idx = tid; idx < LK * (S_SIG / 4); idx += 256) {
            int row = idx >> 2, c4 = idx & 3;
            *(v4h*)(SKh + row * 32 + c4 * 4)      = pack4(gsk[idx]);
            *(v4h*)(SKh + row * 32 + 16 + c4 * 4) = (v4h){};
        }
        for (int r = tid; r < LQ; r += 256) {
            const float* p = sig_q + ((size_t)b * LQ + r) * S_SIG;
            float s = 0.f;
#pragma unroll
            for (int i = 0; i < S_SIG; ++i) s += p[i] * p[i];
            NQ[r] = s;
            RQ[r] = rsqrtf(size_q[(size_t)b * LQ + r]);
        }
        for (int r = tid; r < LK; r += 256) {
            const float* p = sig_k + ((size_t)b * LK + r) * S_SIG;
            float s = 0.f;
#pragma unroll
            for (int i = 0; i < S_SIG; ++i) s += p[i] * p[i];
            NK[r] = s;
            RK[r] = rsqrtf(size_k[(size_t)b * LK + r]);
        }
    }
    __syncthreads();

    // ---------------- Phase 2: q = (phi_q @ W_A^T) * QSCALE -> QB -----------
    {
        // A-frags for this wave's strip are invariant across output tiles:
        // load the 8 K-step fragments once (64 VGPRs), reuse for all 4 nt.
        v16h afr[8];
#pragma unroll
        for (int kk = 0; kk < 8; ++kk)
            afr[kk] = ldsFrag(PH, D_MODEL, m0, 32 * kk, lane);
#pragma unroll
        for (int nt = 0; nt < 4; ++nt) {
            v8f c = {};
#pragma unroll
            for (int kk = 0; kk < 8; ++kk) {
                v16h bf = ldsFrag(W1, D_MODEL, 16 * nt, 32 * kk, lane);
                c = wmma_f16(afr[kk], bf, c);
            }
#pragma unroll
            for (int j = 0; j < 8; ++j)
                QB[(m0 + j + 8 * lh) * DH + 16 * nt + l16] =
                    (_Float16)(c[j] * QSCALE);
        }
    }
    __syncthreads();

    // ---------------- Phase 3: k, v (shared A frags) in 2 key chunks --------
    for (int ch = 0; ch < 2; ++ch) {
        if (ch == 0) {
            const float4* gb = (const float4*)W_B + (size_t)h * DH * (D_MODEL / 4);
            const float4* gv = (const float4*)W_V + (size_t)h * DH * (D_MODEL / 4);
            for (int idx = tid; idx < DH * (D_MODEL / 4); idx += 256) {
                ((v4h*)W1)[idx] = pack4(gb[idx]);
                ((v4h*)W2)[idx] = pack4(gv[idx]);
            }
        }
        const float4* gp = (const float4*)phi_k +
                           ((size_t)b * LK + 128 * ch) * (D_MODEL / 4);
        for (int idx = tid; idx < 128 * (D_MODEL / 4); idx += 256)
            ((v4h*)PH)[idx] = pack4(gp[idx]);
        __syncthreads();

        v16h afr[8];
#pragma unroll
        for (int kk = 0; kk < 8; ++kk)
            afr[kk] = ldsFrag(PH, D_MODEL, m0, 32 * kk, lane);
#pragma unroll
        for (int nt = 0; nt < 4; ++nt) {
            v8f ck = {}, cv = {};
#pragma unroll
            for (int kk = 0; kk < 8; ++kk) {
                v16h bk = ldsFrag(W1, D_MODEL, 16 * nt, 32 * kk, lane);
                v16h bv = ldsFrag(W2, D_MODEL, 16 * nt, 32 * kk, lane);
                ck = wmma_f16(afr[kk], bk, ck);
                cv = wmma_f16(afr[kk], bv, cv);
            }
            // k: row-major [key][dh]
#pragma unroll
            for (int j = 0; j < 8; ++j)
                KB[(128 * ch + m0 + j + 8 * lh) * DH + 16 * nt + l16] =
                    (_Float16)ck[j];
            // v: transposed [dh][key] (contiguous 16B store per lane)
            v8h pk;
#pragma unroll
            for (int j = 0; j < 8; ++j) pk[j] = (_Float16)cv[j];
            *(v8h*)(VT + (16 * nt + l16) * LK + 128 * ch + m0 + 8 * lh) = pk;
        }
        __syncthreads();
    }

    // ---------------- Phase 4: attention (wave-private strips) --------------
    {
        float nqj[8], rqj[8];
#pragma unroll
        for (int j = 0; j < 8; ++j) {
            int r = m0 + j + 8 * lh;
            nqj[j] = NQ[r];
            rqj[j] = RQ[r];
        }
        v16h aq0 = ldsFrag(QB, DH, m0, 0, lane);
        v16h aq1 = ldsFrag(QB, DH, m0, 32, lane);
        v16h asg = ldsFrag(SQh, 32, m0, 0, lane);

        v8f o0 = {}, o1 = {}, o2 = {}, o3 = {};
        float mrow[8], lrow[8];
#pragma unroll
        for (int j = 0; j < 8; ++j) { mrow[j] = -1e30f; lrow[j] = 0.f; }

        for (int ch = 0; ch < 2; ++ch) {
            v8f s[8];
#pragma unroll
            for (int nt = 0; nt < 8; ++nt) {
                const int kt = 8 * ch + nt;
                // signature cross term (K=16 padded to 32)
                v16h bs = ldsFrag(SKh, 32, 16 * kt, 0, lane);
                v8f cc = {};
                cc = wmma_f16(asg, bs, cc);
                // cc := -GE*(nq + nk - 2*cross)*rq*rk  (the delta term)
                float nkc = NK[16 * kt + l16];
                float t1  = GE * RK[16 * kt + l16];
#pragma unroll
                for (int j = 0; j < 8; ++j) {
                    float g = rqj[j] * t1;
                    cc[j] = (2.f * g) * cc[j] - (nqj[j] + nkc) * g;
                }
                // + scaled q.k^T (scale folded into q)
                v16h b0 = ldsFrag(KB, DH, 16 * kt, 0, lane);
                v16h b1 = ldsFrag(KB, DH, 16 * kt, 32, lane);
                cc = wmma_f16(aq0, b0, cc);
                cc = wmma_f16(aq1, b1, cc);
                s[nt] = cc;
            }
            // ---- online softmax over this 128-key chunk ----
            float pm[8];
#pragma unroll
            for (int j = 0; j < 8; ++j) pm[j] = s[0][j];
#pragma unroll
            for (int nt = 1; nt < 8; ++nt)
#pragma unroll
                for (int j = 0; j < 8; ++j) pm[j] = fmaxf(pm[j], s[nt][j]);
#pragma unroll
            for (int mask = 1; mask <= 8; mask <<= 1)
#pragma unroll
                for (int j = 0; j < 8; ++j)
                    pm[j] = fmaxf(pm[j], __shfl_xor(pm[j], mask, 32));
            float sc[8];
#pragma unroll
            for (int j = 0; j < 8; ++j) {
                float nm = fmaxf(mrow[j], pm[j]);
                sc[j] = __expf(mrow[j] - nm);
                mrow[j] = nm;
            }
            float ps[8];
#pragma unroll
            for (int j = 0; j < 8; ++j) ps[j] = 0.f;
#pragma unroll
            for (int nt = 0; nt < 8; ++nt)
#pragma unroll
                for (int j = 0; j < 8; ++j) {
                    s[nt][j] = __expf(s[nt][j] - mrow[j]);
                    ps[j] += s[nt][j];
                }
#pragma unroll
            for (int mask = 1; mask <= 8; mask <<= 1)
#pragma unroll
                for (int j = 0; j < 8; ++j)
                    ps[j] += __shfl_xor(ps[j], mask, 32);
#pragma unroll
            for (int j = 0; j < 8; ++j) {
                lrow[j] = lrow[j] * sc[j] + ps[j];
                o0[j] *= sc[j]; o1[j] *= sc[j];
                o2[j] *= sc[j]; o3[j] *= sc[j];
            }
            // stage P (C-layout -> row-major f16) in this wave's private strip
#pragma unroll
            for (int nt = 0; nt < 8; ++nt)
#pragma unroll
                for (int j = 0; j < 8; ++j)
                    PH[(m0 + j + 8 * lh) * LK + 128 * ch + 16 * nt + l16] =
                        (_Float16)s[nt][j];
            // O += P @ V   (A from PH strip, B from transposed V)
#pragma unroll
            for (int kk = 0; kk < 4; ++kk) {
                v16h ap  = ldsFrag(PH, LK, m0, 128 * ch + 32 * kk, lane);
                v16h bv0 = ldsFrag(VT, LK, 0,  128 * ch + 32 * kk, lane);
                v16h bv1 = ldsFrag(VT, LK, 16, 128 * ch + 32 * kk, lane);
                v16h bv2 = ldsFrag(VT, LK, 32, 128 * ch + 32 * kk, lane);
                v16h bv3 = ldsFrag(VT, LK, 48, 128 * ch + 32 * kk, lane);
                o0 = wmma_f16(ap, bv0, o0);
                o1 = wmma_f16(ap, bv1, o1);
                o2 = wmma_f16(ap, bv2, o2);
                o3 = wmma_f16(ap, bv3, o3);
            }
        }
        // ---- normalize and write context ----
#pragma unroll
        for (int j = 0; j < 8; ++j) {
            float iv = 1.f / lrow[j];
            size_t ro = ((size_t)b * LQ + m0 + j + 8 * lh) * (size_t)D_MODEL +
                        (size_t)h * DH;
            out[ro + l16]      = o0[j] * iv;
            out[ro + 16 + l16] = o1[j] * iv;
            out[ro + 32 + l16] = o2[j] * iv;
            out[ro + 48 + l16] = o3[j] * iv;
        }
    }
}

extern "C" void kernel_launch(void* const* d_in, const int* in_sizes, int n_in,
                              void* d_out, int out_size, void* d_ws, size_t ws_size,
                              hipStream_t stream) {
    const float* phi_q  = (const float*)d_in[0];
    const float* sig_q  = (const float*)d_in[1];
    const float* size_q = (const float*)d_in[2];
    const void*  q_ptrs = d_in[3];
    const float* phi_k  = (const float*)d_in[4];
    const float* sig_k  = (const float*)d_in[5];
    const float* size_k = (const float*)d_in[6];
    // d_in[7] = k_ptrs (unused by the computation)
    const float* W_A    = (const float*)d_in[8];
    const float* W_B    = (const float*)d_in[9];
    const float* W_V    = (const float*)d_in[10];
    float* out = (float*)d_out;

    (void)d_ws; (void)ws_size; (void)n_in;

    hipFuncSetAttribute(reinterpret_cast<const void*>(&setbank_attn_kernel),
                        hipFuncAttributeMaxDynamicSharedMemorySize, SMEM_BYTES);

    dim3 grid(BATCH * NHEAD);   // 2048 workgroups, one per (batch, head)
    setbank_attn_kernel<<<grid, 256, SMEM_BYTES, stream>>>(
        phi_q, sig_q, size_q, phi_k, sig_k, size_k, W_A, W_B, W_V, out);

    // passthrough output: q_ptrs appended after the context tensor
    hipMemcpyAsync(out + (size_t)BATCH * LQ * D_MODEL, q_ptrs,
                   (size_t)in_sizes[3] * sizeof(int),
                   hipMemcpyDeviceToDevice, stream);
}
